// Encoder_3444563771502
// MI455X (gfx1250) — matrix operand order repaired and checked by reading, not compiled, not run on hardware
//
#include <hip/hip_runtime.h>
#include <math.h>

// ---------------- problem constants (from reference) ----------------
#define Dn   1024
#define Hn   16
#define Ln   6
#define Fn   4096
#define Bn   4
#define Sn   2048
#define DKn  64
#define Mrows (Bn * Sn)   // 8192 token rows

// ---------------- WMMA vector types (gfx1250, wave32) ----------------
typedef __attribute__((ext_vector_type(16))) __bf16 v16bf;
typedef __attribute__((ext_vector_type(8)))  float  v8f;

union FragAB {            // one 16x32 (A) or 32x16 (B) bf16 fragment: 8 VGPRs
  v16bf v;
  unsigned int d[8];
};

__device__ __forceinline__ unsigned short f2bf(float f) {
  union { float f; unsigned u; } x; x.f = f;
  unsigned r = x.u + 0x7FFFu + ((x.u >> 16) & 1u);   // round-to-nearest-even
  return (unsigned short)(r >> 16);
}

// =====================================================================
// 1) token embedding * sqrt(D) + sinusoidal positional encoding
// =====================================================================
__global__ void __launch_bounds__(256)
embed_pe_kernel(const int* __restrict__ x, const float* __restrict__ emb,
                float* __restrict__ h) {
  const int idx = blockIdx.x * 256 + threadIdx.x;     // over Mrows*Dn
  const int row = idx >> 10;                          // / Dn
  const int d   = idx & (Dn - 1);
  const int s   = row & (Sn - 1);                     // position within sequence
  const int tok = x[row];
  const float e = emb[(size_t)tok * Dn + d] * 32.0f;  // sqrt(1024)
  const int   eb  = d & ~1;
  const float div = __expf((float)eb * (-9.210340371976184f / (float)Dn));
  const float ang = (float)s * div;
  const float pe  = (d & 1) ? __cosf(ang) : __sinf(ang);
  h[idx] = e + pe;
}

// =====================================================================
// 2) bf16 WMMA GEMM, double-buffered LDS:
//    out[M,N] = A[M,K] * W[K,N] + bias (+resid) (+ReLU)
//    out_mode: 0 -> fp32 out, 1 -> bf16 out (for Q/K/V feeding attention)
//    block = 256 threads (8 waves), tile 128x128, BK=32
// =====================================================================
#define BM 128
#define BN 128
#define BK 32
#define ASTR 34   // ushort stride (pad vs bank conflicts, keeps dword align)
#define BSTR 34

__global__ void __launch_bounds__(256)
gemm_bf16_kernel(const float* __restrict__ A, const float* __restrict__ W,
                 const float* __restrict__ bias, const float* __restrict__ resid,
                 void* __restrict__ outp, int N, int K, int relu, int out_mode) {
  __shared__ unsigned short As[2][BM * ASTR];   // [m][k] bf16
  __shared__ unsigned short Bs[2][BN * BSTR];   // [n][k] bf16 (W transposed)

  const int tid  = threadIdx.x;
  const int lane = tid & 31;
  const int w    = tid >> 5;
  const int m0   = blockIdx.y * BM;
  const int n0   = blockIdx.x * BN;
  const int mbase = (w >> 1) * 32;           // wave's 32-row strip
  const int nbase = (w & 1) * 64;            // wave's 64-col strip
  const int ml   = lane & 15;
  const int kbA  = (lane < 16) ? 0 : 8;      // A-frag K base (ISA 16-bit A layout)
  const int kbB  = (lane < 16) ? 0 : 8;      // B-frag K base (ISA 16-bit B layout)

  const v8f vzero = {0.f,0.f,0.f,0.f,0.f,0.f,0.f,0.f};
  v8f acc[2][4];
  #pragma unroll
  for (int i = 0; i < 2; ++i)
    #pragma unroll
    for (int j = 0; j < 4; ++j) acc[i][j] = vzero;

  float4 ar[4], wr[4];                       // staging registers
  auto gload = [&](int k0) {
    #pragma unroll
    for (int p = 0; p < 4; ++p) {
      ar[p] = *(const float4*)(A + (size_t)(m0 + (tid >> 3) + p * 32) * K +
                               k0 + ((tid & 7) << 2));
      wr[p] = *(const float4*)(W + (size_t)(k0 + (tid >> 5) + p * 8) * N +
                               n0 + ((tid & 31) << 2));
    }
  };
  auto lstore = [&](int buf) {
    #pragma unroll
    for (int p = 0; p < 4; ++p) {
      const int r = (tid >> 3) + p * 32;
      const int c = (tid & 7) << 2;
      unsigned short* dst = &As[buf][r * ASTR + c];
      dst[0] = f2bf(ar[p].x); dst[1] = f2bf(ar[p].y);
      dst[2] = f2bf(ar[p].z); dst[3] = f2bf(ar[p].w);
      const int kr = (tid >> 5) + p * 8;
      const int nc = (tid & 31) << 2;
      Bs[buf][(nc + 0) * BSTR + kr] = f2bf(wr[p].x);
      Bs[buf][(nc + 1) * BSTR + kr] = f2bf(wr[p].y);
      Bs[buf][(nc + 2) * BSTR + kr] = f2bf(wr[p].z);
      Bs[buf][(nc + 3) * BSTR + kr] = f2bf(wr[p].w);
    }
  };

  const int nK = K / BK;
  gload(0);
  lstore(0);
  __syncthreads();

  for (int ks = 0; ks < nK; ++ks) {
    if (ks + 1 < nK) gload((ks + 1) * BK);   // in flight during WMMA burst

    const unsigned short* Ab = As[ks & 1];
    const unsigned short* Bb = Bs[ks & 1];

    FragAB afr[2];
    #pragma unroll
    for (int im = 0; im < 2; ++im) {
      const unsigned short* rowp = &Ab[(mbase + im * 16 + ml) * ASTR];
      #pragma unroll
      for (int j = 0; j < 8; ++j) {
        const int kk = ((j >> 2) << 4) + kbA + ((j & 3) << 1);
        afr[im].d[j] = *(const unsigned int*)(rowp + kk);
      }
    }
    #pragma unroll
    for (int in_ = 0; in_ < 4; ++in_) {
      FragAB bfr;
      const unsigned short* rowp = &Bb[(nbase + in_ * 16 + ml) * BSTR + kbB];
      #pragma unroll
      for (int j = 0; j < 8; ++j) bfr.d[j] = *(const unsigned int*)(rowp + ((j >> 2) << 4) + ((j & 3) << 1));
      #pragma unroll
      for (int im = 0; im < 2; ++im)
        acc[im][in_] = __builtin_amdgcn_wmma_f32_16x16x32_bf16(
            false, afr[im].v, false, bfr.v, (short)0, acc[im][in_], false, false);
    }

    if (ks + 1 < nK) lstore((ks + 1) & 1);   // writes the *other* buffer
    __syncthreads();
  }

  // ---- epilogue: bias + optional residual + optional ReLU ----
  const int mofs = (lane >> 4) << 3;         // C layout: lanes 16-31 hold M+8
  float*          outf = (float*)outp;
  unsigned short* outb = (unsigned short*)outp;
  #pragma unroll
  for (int im = 0; im < 2; ++im) {
    #pragma unroll
    for (int in_ = 0; in_ < 4; ++in_) {
      const int n  = n0 + nbase + in_ * 16 + ml;
      const float bv = bias[n];
      #pragma unroll
      for (int r = 0; r < 8; ++r) {
        const int m = m0 + mbase + im * 16 + mofs + r;
        float val = acc[im][in_][r] + bv;
        if (resid) val += resid[(size_t)m * N + n];
        if (relu)  val = fmaxf(val, 0.f);
        if (out_mode) outb[(size_t)m * N + n] = f2bf(val);
        else          outf[(size_t)m * N + n] = val;
      }
    }
  }
}

// =====================================================================
// 3) flash attention: one workgroup per (b*H + h, query-tile of 64)
//    q/k/v are bf16 [B,S,H,DK]; Q/K tiles staged with async-to-LDS copies
//    (ASYNCcnt path), V transposed in flight; O accumulated in registers.
// =====================================================================
#define QSTR 72   // ushort stride: 144B rows -> 16B-aligned B128 async targets

__global__ void __launch_bounds__(256)
attn_kernel(const unsigned short* __restrict__ q,
            const unsigned short* __restrict__ k,
            const unsigned short* __restrict__ v, float* __restrict__ o) {
  __shared__ __align__(16) unsigned short Qs[64 * QSTR]; // [qrow][dk]
  __shared__ __align__(16) unsigned short Ks[64 * QSTR]; // [key][dk] == B^T layout
  __shared__ unsigned short Vt[64 * 66];                 // [dk][key] == B^T layout
  __shared__ unsigned short Ps[64 * 66];                 // [qrow][key] softmax num.
  __shared__ float Sc[64 * 65];                          // scores fp32
  __shared__ float mrow[64], lrow[64], arow[64];

  const int bh  = blockIdx.x;
  const int bb  = bh / Hn, hh = bh % Hn;
  const int qt  = blockIdx.y;
  const int tid = threadIdx.x;
  const int lane = tid & 31, w = tid >> 5;
  const int mt   = w >> 1;                 // wave's 16-row m-tile (0..3)
  const int ntA  = (w & 1) << 1;           // wave's first n-tile (0 or 2)
  const int ml   = lane & 15;
  const int kbA  = (lane < 16) ? 0 : 8;
  const int kbB  = (lane < 16) ? 0 : 8;
  const int mofs = (lane >> 4) << 3;

  const size_t base_e = ((size_t)(bb * Sn + qt * 64)) * Dn + hh * DKn;

  // ---- Q tile: pure bf16 copy -> async load to LDS (gfx1250 ASYNC path) ----
  #pragma unroll
  for (int p = 0; p < 2; ++p) {
    const int r  = (tid >> 3) + p * 32;        // 0..63
    const int c8 = (tid & 7) << 3;             // 0,8,..,56 (8 bf16 = 16B)
    const unsigned lds_off = (unsigned)(size_t)&Qs[r * QSTR + c8];
    const unsigned short* gptr = q + base_e + (size_t)r * Dn + c8;
    asm volatile("global_load_async_to_lds_b128 %0, %1, off"
                 :: "v"(lds_off), "v"(gptr) : "memory");
  }
  if (tid < 64) { mrow[tid] = -3.0e38f; lrow[tid] = 0.f; }
  asm volatile("s_wait_asynccnt 0x0" ::: "memory");
  __syncthreads();

  FragAB qa[2];                            // Q fragments, kk = 0..31 / 32..63
  {
    const unsigned short* rowp = &Qs[(mt * 16 + ml) * QSTR];
    #pragma unroll
    for (int f = 0; f < 2; ++f)
      #pragma unroll
      for (int j = 0; j < 8; ++j) {
        const int kk = f * 32 + ((j >> 2) << 4) + kbA + ((j & 3) << 1);
        qa[f].d[j] = *(const unsigned int*)(rowp + kk);
      }
  }

  const v8f vzero = {0.f,0.f,0.f,0.f,0.f,0.f,0.f,0.f};
  v8f oacc[2] = {vzero, vzero};

  for (int kt = 0; kt < Sn / 64; ++kt) {
    const size_t basek = ((size_t)(bb * Sn + kt * 64)) * Dn + hh * DKn;
    #pragma unroll
    for (int p = 0; p < 2; ++p) {
      const int r  = (tid >> 3) + p * 32;
      const int c8 = (tid & 7) << 3;
      // K tile: async bf16 copy into LDS (row-major == transposed-B layout)
      const unsigned lds_off = (unsigned)(size_t)&Ks[r * QSTR + c8];
      const unsigned short* gptr = k + basek + (size_t)r * Dn + c8;
      asm volatile("global_load_async_to_lds_b128 %0, %1, off"
                   :: "v"(lds_off), "v"(gptr) : "memory");
      // V tile: load 8 bf16 and transpose-scatter into Vt[dk][key]
      const uint4 vv = *(const uint4*)(v + basek + (size_t)r * Dn + c8);
      Vt[(c8 + 0) * 66 + r] = (unsigned short)(vv.x & 0xffffu);
      Vt[(c8 + 1) * 66 + r] = (unsigned short)(vv.x >> 16);
      Vt[(c8 + 2) * 66 + r] = (unsigned short)(vv.y & 0xffffu);
      Vt[(c8 + 3) * 66 + r] = (unsigned short)(vv.y >> 16);
      Vt[(c8 + 4) * 66 + r] = (unsigned short)(vv.z & 0xffffu);
      Vt[(c8 + 5) * 66 + r] = (unsigned short)(vv.z >> 16);
      Vt[(c8 + 6) * 66 + r] = (unsigned short)(vv.w & 0xffffu);
      Vt[(c8 + 7) * 66 + r] = (unsigned short)(vv.w >> 16);
    }
    asm volatile("s_wait_asynccnt 0x0" ::: "memory");
    __syncthreads();

    // ---- S = Q K^T / sqrt(DK) via WMMA ----
    #pragma unroll
    for (int in_ = 0; in_ < 2; ++in_) {
      v8f sacc = vzero;
      #pragma unroll
      for (int f = 0; f < 2; ++f) {
        FragAB bfr;
        const unsigned short* rowp = &Ks[((ntA + in_) * 16 + ml) * QSTR + f * 32 + kbB];
        #pragma unroll
        for (int j = 0; j < 8; ++j) bfr.d[j] = *(const unsigned int*)(rowp + ((j >> 2) << 4) + ((j & 3) << 1));
        sacc = __builtin_amdgcn_wmma_f32_16x16x32_bf16(
            false, qa[f].v, false, bfr.v, (short)0, sacc, false, false);
      }
      const int nn = (ntA + in_) * 16 + ml;
      const int mm = mt * 16 + mofs;
      #pragma unroll
      for (int r = 0; r < 8; ++r) Sc[(mm + r) * 65 + nn] = sacc[r] * 0.125f;
    }
    __syncthreads();

    // ---- online softmax (one thread per query row) ----
    if (tid < 64) {
      const int r = tid;
      const float mold = mrow[r];
      float mx = mold;
      for (int j = 0; j < 64; ++j) mx = fmaxf(mx, Sc[r * 65 + j]);
      float sum = 0.f;
      for (int j = 0; j < 64; ++j) {
        const float p = __expf(Sc[r * 65 + j] - mx);
        sum += p;
        Ps[r * 66 + j] = f2bf(p);
      }
      const float alpha = __expf(mold - mx);
      lrow[r] = lrow[r] * alpha + sum;
      mrow[r] = mx;
      arow[r] = alpha;
    }
    __syncthreads();

    // ---- rescale O accumulators, then O += P V via WMMA ----
    #pragma unroll
    for (int in_ = 0; in_ < 2; ++in_)
      #pragma unroll
      for (int r = 0; r < 8; ++r)
        oacc[in_][r] *= arow[mt * 16 + mofs + r];

    FragAB pa[2];
    {
      const unsigned short* rowp = &Ps[(mt * 16 + ml) * 66];
      #pragma unroll
      for (int f = 0; f < 2; ++f)
        #pragma unroll
        for (int j = 0; j < 8; ++j) {
          const int kk = f * 32 + ((j >> 2) << 4) + kbA + ((j & 3) << 1);
          pa[f].d[j] = *(const unsigned int*)(rowp + kk);
        }
    }
    #pragma unroll
    for (int in_ = 0; in_ < 2; ++in_) {
      #pragma unroll
      for (int f = 0; f < 2; ++f) {
        FragAB bfr;
        const unsigned short* rowp = &Vt[((ntA + in_) * 16 + ml) * 66 + f * 32 + kbB];
        #pragma unroll
        for (int j = 0; j < 8; ++j) bfr.d[j] = *(const unsigned int*)(rowp + ((j >> 2) << 4) + ((j & 3) << 1));
        oacc[in_] = __builtin_amdgcn_wmma_f32_16x16x32_bf16(
            false, pa[f].v, false, bfr.v, (short)0, oacc[in_], false, false);
      }
    }
    __syncthreads();
  }

  // ---- writeout: ctx[b,s,h,dk] = O / l  (fp32 for the O-projection GEMM) ----
  #pragma unroll
  for (int in_ = 0; in_ < 2; ++in_) {
    const int nn = (ntA + in_) * 16 + ml;
    #pragma unroll
    for (int r = 0; r < 8; ++r) {
      const int mm = mt * 16 + mofs + r;
      o[base_e + (size_t)mm * Dn + nn] = oacc[in_][r] / lrow[mm];
    }
  }
}

// =====================================================================
// 4) layernorm: one workgroup (256 threads) per token row
// =====================================================================
__global__ void __launch_bounds__(256)
ln_kernel(const float* __restrict__ in, const float* __restrict__ g,
          const float* __restrict__ b, float* __restrict__ out) {
  __shared__ float red[256];
  const int row = blockIdx.x;
  const int tid = threadIdx.x;
  const float* xr = in + (size_t)row * Dn;
  float s = 0.f;
  for (int i = tid; i < Dn; i += 256) s += xr[i];
  red[tid] = s; __syncthreads();
  for (int off = 128; off > 0; off >>= 1) {
    if (tid < off) red[tid] += red[tid + off];
    __syncthreads();
  }
  const float mu = red[0] * (1.0f / Dn);
  __syncthreads();
  float vv = 0.f;
  for (int i = tid; i < Dn; i += 256) { const float d2 = xr[i] - mu; vv += d2 * d2; }
  red[tid] = vv; __syncthreads();
  for (int off = 128; off > 0; off >>= 1) {
    if (tid < off) red[tid] += red[tid + off];
    __syncthreads();
  }
  const float rstd = rsqrtf(red[0] * (1.0f / Dn) + 1e-5f);
  float* orow = out + (size_t)row * Dn;
  for (int i = tid; i < Dn; i += 256)
    orow[i] = (xr[i] - mu) * rstd * g[i] + b[i];
}

// =====================================================================
// launcher
// =====================================================================
extern "C" void kernel_launch(void* const* d_in, const int* in_sizes, int n_in,
                              void* d_out, int out_size, void* d_ws, size_t ws_size,
                              hipStream_t stream) {
  (void)in_sizes; (void)n_in; (void)out_size; (void)ws_size;
  const int*   x    = (const int*)  d_in[0];
  const float* emb  = (const float*)d_in[1];
  const float* Wq   = (const float*)d_in[2];
  const float* bq   = (const float*)d_in[3];
  const float* Wk   = (const float*)d_in[4];
  const float* bk   = (const float*)d_in[5];
  const float* Wv   = (const float*)d_in[6];
  const float* bv   = (const float*)d_in[7];
  const float* Wo   = (const float*)d_in[8];
  const float* bo   = (const float*)d_in[9];
  const float* W1   = (const float*)d_in[10];
  const float* b1   = (const float*)d_in[11];
  const float* W2   = (const float*)d_in[12];
  const float* b2   = (const float*)d_in[13];
  const float* ln1g = (const float*)d_in[14];
  const float* ln1b = (const float*)d_in[15];
  const float* ln2g = (const float*)d_in[16];
  const float* ln2b = (const float*)d_in[17];

  float* ws = (float*)d_ws;
  const size_t MD = (size_t)Mrows * Dn;
  float* h   = ws;
  unsigned short* qb = (unsigned short*)(ws + 1 * MD);  // bf16 [B,S,H,DK]
  unsigned short* kb = (unsigned short*)(ws + 2 * MD);
  unsigned short* vb = (unsigned short*)(ws + 3 * MD);
  float* ctx = ws + 4 * MD;
  float* tmp = ws + 5 * MD;
  float* ff  = ws + 6 * MD;   // Mrows * Fn floats

  embed_pe_kernel<<<(Mrows * Dn) / 256, 256, 0, stream>>>(x, emb, h);

  const dim3 gD(Dn / BN, Mrows / BM);
  const dim3 gF(Fn / BN, Mrows / BM);
  const dim3 gAttn(Bn * Hn, Sn / 64);

  for (int l = 0; l < Ln; ++l) {
    const float* wq = Wq + (size_t)l * Dn * Dn;
    const float* wk = Wk + (size_t)l * Dn * Dn;
    const float* wv = Wv + (size_t)l * Dn * Dn;
    const float* wo = Wo + (size_t)l * Dn * Dn;
    const float* w1 = W1 + (size_t)l * Dn * Fn;
    const float* w2 = W2 + (size_t)l * Fn * Dn;
    const float* bql = bq + (size_t)l * Dn;
    const float* bkl = bk + (size_t)l * Dn;
    const float* bvl = bv + (size_t)l * Dn;
    const float* bol = bo + (size_t)l * Dn;
    const float* b1l = b1 + (size_t)l * Fn;
    const float* b2l = b2 + (size_t)l * Dn;

    gemm_bf16_kernel<<<gD, 256, 0, stream>>>(h, wq, bql, nullptr, qb, Dn, Dn, 0, 1);
    gemm_bf16_kernel<<<gD, 256, 0, stream>>>(h, wk, bkl, nullptr, kb, Dn, Dn, 0, 1);
    gemm_bf16_kernel<<<gD, 256, 0, stream>>>(h, wv, bvl, nullptr, vb, Dn, Dn, 0, 1);
    attn_kernel<<<gAttn, 256, 0, stream>>>(qb, kb, vb, ctx);
    gemm_bf16_kernel<<<gD, 256, 0, stream>>>(ctx, wo, bol, h, tmp, Dn, Dn, 0, 0);
    ln_kernel<<<Mrows, 256, 0, stream>>>(tmp, ln1g + (size_t)l * Dn,
                                         ln1b + (size_t)l * Dn, h);
    gemm_bf16_kernel<<<gF, 256, 0, stream>>>(h, w1, b1l, nullptr, ff, Fn, Dn, 1, 0);
    gemm_bf16_kernel<<<gD, 256, 0, stream>>>(ff, w2, b2l, h, tmp, Dn, Fn, 0, 0);
    ln_kernel<<<Mrows, 256, 0, stream>>>(tmp, ln2g + (size_t)l * Dn,
                                         ln2b + (size_t)l * Dn, h);
  }

  hipMemcpyAsync(d_out, h, MD * sizeof(float), hipMemcpyDeviceToDevice, stream);
}
